// DCNv3_Up_55207509623209
// MI455X (gfx1250) — matrix operand order repaired and checked by reading, not compiled, not run on hardware
//
#include <hip/hip_runtime.h>
#include <math.h>
#include <stdint.h>

typedef __attribute__((ext_vector_type(16))) _Float16 v16h;
typedef __attribute__((ext_vector_type(8)))  _Float16 v8h;
typedef __attribute__((ext_vector_type(8)))  float    v8f;

// Problem constants
#define NB   4      // batch
#define HIN  48
#define WIN  48
#define CH   256
#define HP   99     // padded/zero-stuffed grid
#define HO   96     // output spatial
#define GR   8
#define PP   16
#define GC   32
#define LDA  264    // LDS A row stride in halves (16B aligned, bank-conflict free)

#define WMMA_F16(a, b, c) __builtin_amdgcn_wmma_f32_16x16x32_f16(false, (a), false, (b), (short)0, (c), false, false)

// ---- CDNA5 async global->LDS copy (ASYNCcnt-tracked, no VGPR round trip) ----
__device__ __forceinline__ void async_copy_b128(uint32_t lds_byte_addr, const void* gaddr) {
    asm volatile("global_load_async_to_lds_b128 %0, %1, off"
                 :: "v"(lds_byte_addr), "v"(gaddr) : "memory");
}
__device__ __forceinline__ void wait_asynccnt0() {
    asm volatile("s_wait_asynccnt 0" ::: "memory");
}

// ---------------- Weight prep: transpose + f32->f16 ----------------
// wt layout: [in 256x256][off 256x256][mask 128x256][out 256x256], row = out-ch n, K contiguous
__global__ void prep_weights(const float* __restrict__ in_w, const float* __restrict__ off_w,
                             const float* __restrict__ mask_w, const float* __restrict__ out_w,
                             _Float16* __restrict__ wt) {
    int t = blockIdx.x * 256 + threadIdx.x;       // 0..229375
    if (t < 65536)      { int n = t >> 8, k = t & 255;               wt[t] = (_Float16)in_w  [k * 256 + n]; }
    else if (t < 131072){ int u = t - 65536;  int n = u >> 8, k = u & 255; wt[t] = (_Float16)off_w [k * 256 + n]; }
    else if (t < 163840){ int u = t - 131072; int n = u >> 8, k = u & 255; wt[t] = (_Float16)mask_w[k * 128 + n]; }
    else                { int u = t - 163840; int n = u >> 8, k = u & 255; wt[t] = (_Float16)out_w [k * 256 + n]; }
}

// ---------------- Fill xq with input_proj bias (zero-stuffed rows) ----------------
__global__ void fill_bias(float* __restrict__ xq, const float* __restrict__ in_b, int total) {
    int idx = blockIdx.x * 256 + threadIdx.x;
    if (idx < total) xq[idx] = in_b[idx & 255];
}

// ---------------- WMMA GEMM: Out[M x Nc] = A[M x 256] * Bt^T + bias ----------------
// Bt is pre-transposed f16 [Nc][256]. mode==1: scatter rows into zero-stuffed xq grid.
// Block = 256 threads (8 waves), covers 32 rows x 256 cols. Each wave: 32x32 register tile
// (2x2 WMMA tiles, each A/B fragment reused twice -> 4 WMMA per 4 LDS + 4 global b128 loads).
__global__ void gemm_wmma(const float* __restrict__ A, const _Float16* __restrict__ Bt,
                          const float* __restrict__ bias, float* __restrict__ Out,
                          int Nc, int mode) {
    __shared__ _Float16 s_a[32 * LDA];

    const int tid  = threadIdx.x;
    const int lane = tid & 31;
    const int wave = tid >> 5;
    const int rowBase = blockIdx.x * 32;

    // Stage 32x256 f32 A-panel into LDS as f16 (coalesced: thread t loads col t of row i)
    #pragma unroll
    for (int i = 0; i < 32; ++i)
        s_a[i * LDA + tid] = (_Float16)A[(size_t)(rowBase + i) * 256 + tid];
    __syncthreads();

    const int colBase = wave * 32;
    if (colBase < Nc) {
        const bool hi = (lane & 16) != 0;
        const int m  = lane & 15;              // A row within 16-row tile
        const int n0 = colBase + (lane & 15);  // first D column
        const int n1 = n0 + 16;                // second D column

        v8f acc[2][2] = {};
        #pragma unroll
        for (int kk = 0; kk < 256; kk += 32) {
            // A fragments (two row tiles): lanes 0-15 K=kk..kk+15; lanes 16-31 K=kk+16..kk+31
            int kb = kk + (hi ? 16 : 0);
            v8h a00 = *(const v8h*)(s_a + m * LDA + kb);
            v8h a01 = *(const v8h*)(s_a + m * LDA + kb + 8);
            v8h a10 = *(const v8h*)(s_a + (m + 16) * LDA + kb);
            v8h a11 = *(const v8h*)(s_a + (m + 16) * LDA + kb + 8);
            // B fragments (two col tiles): lanes 0-15 K=kk..kk+15; lanes 16-31 K=kk+16..kk+31
            int kb2 = kk + (hi ? 16 : 0);
            const v8h* bp0 = (const v8h*)(Bt + (size_t)n0 * 256 + kb2);
            const v8h* bp1 = (const v8h*)(Bt + (size_t)n1 * 256 + kb2);
            v8h b00 = bp0[0], b01 = bp0[1];
            v8h b10 = bp1[0], b11 = bp1[1];
            v16h a0, a1, b0, b1;
            #pragma unroll
            for (int i = 0; i < 8; ++i) {
                a0[i] = a00[i]; a0[i + 8] = a01[i];
                a1[i] = a10[i]; a1[i + 8] = a11[i];
                b0[i] = b00[i]; b0[i + 8] = b01[i];
                b1[i] = b10[i]; b1[i + 8] = b11[i];
            }
            acc[0][0] = WMMA_F16(a0, b0, acc[0][0]);
            acc[0][1] = WMMA_F16(a0, b1, acc[0][1]);
            acc[1][0] = WMMA_F16(a1, b0, acc[1][0]);
            acc[1][1] = WMMA_F16(a1, b1, acc[1][1]);
        }

        const float bv0 = bias[n0];
        const float bv1 = bias[n1];
        #pragma unroll
        for (int rt = 0; rt < 2; ++rt) {
            #pragma unroll
            for (int r = 0; r < 8; ++r) {
                int grow = rowBase + rt * 16 + r + (hi ? 8 : 0);  // global A row
                int orow;
                if (mode == 1) {
                    // scatter into zero-stuffed HP x HP grid: (n,h,w) -> (n, 2h+2, 2w+2)
                    int nb  = grow / (HIN * WIN);
                    int rem = grow - nb * (HIN * WIN);
                    int h = rem / WIN, w = rem - h * WIN;
                    orow = nb * (HP * HP) + (2 * h + 2) * HP + (2 * w + 2);
                } else {
                    orow = grow;
                }
                float* o = Out + (size_t)orow * Nc;
                o[n0] = acc[rt][0][r] + bv0;
                o[n1] = acc[rt][1][r] + bv1;
            }
        }
    }
}

// ---------------- Depthwise 4x4 (on implicit zero-stuffed grid) + LayerNorm + exact GELU ----------------
// One block per output pixel (n,i,j), thread = channel.
__global__ void dw_ln_gelu(const float* __restrict__ x, const float* __restrict__ dw_w,
                           const float* __restrict__ dw_b, const float* __restrict__ ln_g,
                           const float* __restrict__ ln_b, float* __restrict__ y) {
    __shared__ float red[256];
    const int b = blockIdx.x;                 // 0..N*HO*HO-1
    const int c = threadIdx.x;
    const int n = b / (HO * HO);
    const int rem = b - n * (HO * HO);
    const int i = rem / HO, j = rem - (rem / HO) * HO;

    float sum = dw_b[c];
    #pragma unroll
    for (int a = 0; a < 4; ++a) {
        int r = i + a;
        if (((r & 1) == 0) && r >= 2 && r <= 96) {
            int hx = (r - 2) >> 1;
            #pragma unroll
            for (int bb = 0; bb < 4; ++bb) {
                int s = j + bb;
                if (((s & 1) == 0) && s >= 2 && s <= 96) {
                    int wx = (s - 2) >> 1;
                    sum += x[(((size_t)n * HIN + hx) * WIN + wx) * CH + c] * dw_w[(a * 4 + bb) * CH + c];
                }
            }
        }
    }

    // LayerNorm over 256 channels
    red[c] = sum; __syncthreads();
    for (int s = 128; s > 0; s >>= 1) { if (c < s) red[c] += red[c + s]; __syncthreads(); }
    float mu = red[0] * (1.0f / 256.0f); __syncthreads();
    float d = sum - mu;
    red[c] = d * d; __syncthreads();
    for (int s = 128; s > 0; s >>= 1) { if (c < s) red[c] += red[c + s]; __syncthreads(); }
    float var = red[0] * (1.0f / 256.0f);
    float v = d * rsqrtf(var + 1e-6f) * ln_g[c] + ln_b[c];
    // exact GELU
    v = 0.5f * v * (1.0f + erff(v * 0.70710678118654752f));
    y[(size_t)b * CH + c] = v;
}

// ---------------- Softmax over P=16 per (pixel, group) ----------------
__global__ void softmax16(float* __restrict__ maskb, int ngroups) {
    int idx = blockIdx.x * 256 + threadIdx.x;
    if (idx >= ngroups) return;
    int row = idx >> 3, g = idx & 7;
    float* p = maskb + (size_t)row * (GR * PP) + g * PP;
    float m[PP];
    float mx = -1e30f;
    #pragma unroll
    for (int k = 0; k < PP; ++k) { m[k] = p[k]; mx = fmaxf(mx, m[k]); }
    float s = 0.f;
    #pragma unroll
    for (int k = 0; k < PP; ++k) { m[k] = __expf(m[k] - mx); s += m[k]; }
    float inv = 1.0f / s;
    #pragma unroll
    for (int k = 0; k < PP; ++k) p[k] = m[k] * inv;
}

// ---------------- DCNv3 core: deformable bilinear gather + mask accumulate ----------------
// Block per pixel; thread = (group g = tid>>5, channel-in-group cc = tid&31) -> coalesced gathers.
// Per-pixel offsets/masks staged into LDS via CDNA5 async global->LDS b128 copies.
// Flips folded in: gather from xq[98-y][98-x], store to core[95-ho][95-wo].
__global__ void dcn_core(const float* __restrict__ xq, const float* __restrict__ offb,
                         const float* __restrict__ maskb, float* __restrict__ coreb) {
    __shared__ __attribute__((aligned(16))) float s_off[256];
    __shared__ __attribute__((aligned(16))) float s_m[128];
    const int b = blockIdx.x;
    const int tid = threadIdx.x;
    const int n = b / (HO * HO);
    const int rem = b - n * (HO * HO);
    const int ho = rem / HO, wo = rem - (rem / HO) * HO;
    const int g = tid >> 5, cc = tid & 31;

    // Async-stage 1KB of offsets (threads 0..63) and 512B of masks (threads 64..95), 16B each.
    if (tid < 64) {
        async_copy_b128((uint32_t)(uintptr_t)&s_off[tid * 4], offb + (size_t)b * 256 + tid * 4);
    } else if (tid < 96) {
        int t = tid - 64;
        async_copy_b128((uint32_t)(uintptr_t)&s_m[t * 4], maskb + (size_t)b * 128 + t * 4);
    }
    wait_asynccnt0();
    __syncthreads();

    const float* xg = xq + ((size_t)n * HP * HP) * CH + g * GC + cc;
    float acc = 0.f;
    #pragma unroll
    for (int p = 0; p < PP; ++p) {
        float o0 = s_off[g * 32 + p * 2];
        float o1 = s_off[g * 32 + p * 2 + 1];
        float px = (float)(wo + (p >> 2)) + o0;        // wo + 1 + lin[p>>2] + off; lin = (p>>2)-1
        float py = (float)(ho + (p & 3)) + o1;
        float fx = floorf(px), fy = floorf(py);
        float tx = px - fx, ty = py - fy;
        int x0 = (int)fx, y0 = (int)fy;
        float w00 = (1.f - tx) * (1.f - ty), w10 = tx * (1.f - ty);
        float w01 = (1.f - tx) * ty,         w11 = tx * ty;
        float mk = s_m[g * 16 + p];
        #pragma unroll
        for (int corner = 0; corner < 4; ++corner) {
            int xi = x0 + (corner & 1), yi = y0 + (corner >> 1);
            float w = (corner == 0) ? w00 : (corner == 1) ? w10 : (corner == 2) ? w01 : w11;
            bool valid = (xi >= 0) & (xi < HP) & (yi >= 0) & (yi < HP);
            int xc = min(max(xi, 0), HP - 1), yc = min(max(yi, 0), HP - 1);
            float v = xg[((size_t)(98 - yc) * HP + (98 - xc)) * CH];
            acc += mk * w * (valid ? v : 0.f);
        }
    }
    coreb[(((size_t)n * HO + (95 - ho)) * HO + (95 - wo)) * CH + g * GC + cc] = acc;
}

// ---------------- Launch ----------------
extern "C" void kernel_launch(void* const* d_in, const int* in_sizes, int n_in,
                              void* d_out, int out_size, void* d_ws, size_t ws_size,
                              hipStream_t stream) {
    const float* x      = (const float*)d_in[0];
    const float* dw_w   = (const float*)d_in[1];
    const float* dw_b   = (const float*)d_in[2];
    const float* ln_g   = (const float*)d_in[3];
    const float* ln_b   = (const float*)d_in[4];
    const float* off_w  = (const float*)d_in[5];
    const float* off_b  = (const float*)d_in[6];
    const float* mask_w = (const float*)d_in[7];
    const float* mask_b = (const float*)d_in[8];
    const float* in_w   = (const float*)d_in[9];
    const float* in_b   = (const float*)d_in[10];
    const float* out_w  = (const float*)d_in[11];
    const float* out_b  = (const float*)d_in[12];
    float* out = (float*)d_out;

    char* ws = (char*)d_ws;
    const size_t XQ_ELEMS   = (size_t)NB * HP * HP * CH;   // 10,036,224
    const size_t Y_ELEMS    = (size_t)NB * HO * HO * CH;   //  9,437,184
    const size_t MASK_ELEMS = (size_t)NB * HO * HO * 128;  //  4,718,592
    float* xq    = (float*)ws;                                 ws += XQ_ELEMS * 4;
    float* ybuf  = (float*)ws;                                 ws += Y_ELEMS * 4;   // reused as coreb
    float* offb  = (float*)ws;                                 ws += Y_ELEMS * 4;
    float* maskb = (float*)ws;                                 ws += MASK_ELEMS * 4;
    _Float16* wt = (_Float16*)ws;                              // 229,376 halves
    _Float16* wt_in   = wt;
    _Float16* wt_off  = wt + 65536;
    _Float16* wt_mask = wt + 131072;
    _Float16* wt_out  = wt + 163840;
    float* coreb = ybuf;  // y dead after offset/mask projections

    prep_weights<<<896, 256, 0, stream>>>(in_w, off_w, mask_w, out_w, wt);
    fill_bias<<<(int)((XQ_ELEMS + 255) / 256), 256, 0, stream>>>(xq, in_b, (int)XQ_ELEMS);

    // input_proj over the 9216 nonzero rows, scattered into the zero-stuffed grid
    gemm_wmma<<<dim3(9216 / 32, 1), 256, 0, stream>>>(x, wt_in, in_b, xq, 256, 1);

    dw_ln_gelu<<<NB * HO * HO, 256, 0, stream>>>(x, dw_w, dw_b, ln_g, ln_b, ybuf);

    gemm_wmma<<<dim3((NB * HO * HO) / 32, 1), 256, 0, stream>>>(ybuf, wt_off, off_b, offb, 256, 0);
    gemm_wmma<<<dim3((NB * HO * HO) / 32, 1), 256, 0, stream>>>(ybuf, wt_mask, mask_b, maskb, 128, 0);

    softmax16<<<(NB * HO * HO * GR + 255) / 256, 256, 0, stream>>>(maskb, NB * HO * HO * GR);

    dcn_core<<<NB * HO * HO, 256, 0, stream>>>(xq, offb, maskb, coreb);

    gemm_wmma<<<dim3((NB * HO * HO) / 32, 1), 256, 0, stream>>>(coreb, wt_out, out_b, out, 256, 0);
}
